// SoftHeadComputer_86371792323238
// MI455X (gfx1250) — matrix operand order repaired and checked by reading, not compiled, not run on hardware
//
#include <hip/hip_runtime.h>

typedef __attribute__((ext_vector_type(16))) _Float16 v16h;
typedef __attribute__((ext_vector_type(8)))  float    v8f;

#define B_    16
#define D_    768
#define L_    96
#define SPANS (L_ * L_)   // 9216
#define KTOP  3

__constant__ float c_posw[6] = {2.0f, 1.5f, 1.0f, 0.8f, 0.3f, 0.5f};
__constant__ float c_depw[6] = {2.0f, 1.5f, 1.5f, 1.2f, 1.0f, 0.5f};

__device__ __forceinline__ v16h load_b_frag(const float* __restrict__ p) {
    // 16 consecutive f32 (K-contiguous in token_embeds), convert to f16.
    v16h v;
    const float4* p4 = (const float4*)p;   // 16B aligned by construction
#pragma unroll
    for (int q = 0; q < 4; ++q) {
        float4 x = p4[q];
        v[q * 4 + 0] = (_Float16)x.x;
        v[q * 4 + 1] = (_Float16)x.y;
        v[q * 4 + 2] = (_Float16)x.z;
        v[q * 4 + 3] = (_Float16)x.w;
    }
    return v;
}

__global__ __launch_bounds__(256)
void span_softhead_kernel(const float* __restrict__ te,   // (B, D, L)
                          const int*   __restrict__ dep,  // (B, L)
                          const int*   __restrict__ pos,  // (B, L)
                          const int*   __restrict__ rel,  // (B, L)
                          float* __restrict__ out)
{
    const int st   = blockIdx.x;        // span tile: 16 spans, 0..575
    const int b    = blockIdx.y;        // batch
    const int tid  = threadIdx.x;
    const int lane = tid & 31;
    const int wave = tid >> 5;          // 0..7
    const int hsel = lane >> 4;         // 0 or 1 (half-wave)
    const int l16  = lane & 15;

    __shared__ float    s_base[L_];
    __shared__ _Float16 s_probs[16 * L_];            // 16 spans x 96 tokens, f16
    __shared__ __align__(16) float s_stage[8][16][36]; // per-wave 16x32 store stage (+pad)

    // ---------------- Phase 1: per-token base score ----------------
    if (tid < L_) {
        const int t = tid;
        int deg = 0;
#pragma unroll 4
        for (int p = 0; p < L_; ++p) deg += (dep[b * L_ + p] == t) ? 1 : 0;
        s_base[t] = 2.0f + 1.5f * (float)deg
                  + 0.8f * c_posw[pos[b * L_ + t]]
                  + 0.5f * c_depw[rel[b * L_ + t]];
    }
    __syncthreads();

    float* hw = out + (size_t)B_ * SPANS * D_;            // head_weights
    float* hi = hw  + (size_t)B_ * SPANS * KTOP;          // head_indices (as float)

    // ------------- Phase 2: softmax + top-3, 2 spans per wave -------------
    for (int j = 0; j < 2; ++j) {
        const int rho = wave * 2 + j;            // local span row 0..15
        const int s   = st * 16 + rho;           // global span index = l*96 + r
        const int l   = s / L_;
        const int r   = s % L_;
        const int t0 = lane, t1 = lane + 32, t2 = lane + 64;

        if (l <= r) {
            const float center = 0.5f * (float)(l + r);
            const float invT = 1.0f / 0.7f;
            float sc0 = (t0 >= l && t0 <= r)
                      ? (s_base[t0] + 0.5f / (1.0f + fabsf((float)t0 - center))) * invT : 0.0f;
            float sc1 = (t1 >= l && t1 <= r)
                      ? (s_base[t1] + 0.5f / (1.0f + fabsf((float)t1 - center))) * invT : 0.0f;
            float sc2 = (t2 >= l && t2 <= r)
                      ? (s_base[t2] + 0.5f / (1.0f + fabsf((float)t2 - center))) * invT : 0.0f;

            float m = fmaxf(sc0, fmaxf(sc1, sc2));
#pragma unroll
            for (int off = 16; off; off >>= 1) m = fmaxf(m, __shfl_xor(m, off, 32));

            float e0 = __expf(sc0 - m), e1 = __expf(sc1 - m), e2 = __expf(sc2 - m);
            float z = e0 + e1 + e2;
#pragma unroll
            for (int off = 16; off; off >>= 1) z += __shfl_xor(z, off, 32);
            const float inv = 1.0f / z;
            const float p0 = e0 * inv, p1 = e1 * inv, p2 = e2 * inv;

            s_probs[rho * L_ + t0] = (_Float16)p0;
            s_probs[rho * L_ + t1] = (_Float16)p1;
            s_probs[rho * L_ + t2] = (_Float16)p2;

            // top-3 with jax tie-break (lower index wins on equal value)
            float v0 = p0, v1 = p1, v2 = p2;
#pragma unroll
            for (int k = 0; k < KTOP; ++k) {
                float bv = v0; int bi = t0;
                if (v1 > bv) { bv = v1; bi = t1; }   // t1 > t0: strict > keeps lower idx
                if (v2 > bv) { bv = v2; bi = t2; }
#pragma unroll
                for (int off = 16; off; off >>= 1) {
                    float ov = __shfl_xor(bv, off, 32);
                    int   oi = __shfl_xor(bi, off, 32);
                    if (ov > bv || (ov == bv && oi < bi)) { bv = ov; bi = oi; }
                }
                if (lane == 0) {
                    hw[((size_t)b * SPANS + s) * KTOP + k] = bv;
                    hi[((size_t)b * SPANS + s) * KTOP + k] = (float)bi;
                }
                if      (bi == t0) v0 = -1.0f;
                else if (bi == t1) v1 = -1.0f;
                else if (bi == t2) v2 = -1.0f;
            }
        } else {
            // invalid span: zero probs row -> GEMM writes zeros; weights 0, indices -1
            s_probs[rho * L_ + t0] = (_Float16)0.0f;
            s_probs[rho * L_ + t1] = (_Float16)0.0f;
            s_probs[rho * L_ + t2] = (_Float16)0.0f;
            if (lane < KTOP) {
                hw[((size_t)b * SPANS + s) * KTOP + lane] = 0.0f;
                hi[((size_t)b * SPANS + s) * KTOP + lane] = -1.0f;
            }
        }
    }
    __syncthreads();

    // ------------- Phase 3: WMMA GEMM, (16 spans x 96) @ (96 x 768) -------------
    // A fragments (K = 0..31, 32..63, 64..95), built once per wave from LDS.
    // 16-bit A 16x32 layout: lanes 0-15: K = 0-15; lanes 16-31: K = 16-31 (matches B).
    v16h a0, a1, a2;
    {
        const int rowoff = l16 * L_;
        const int kofs = hsel * 16;
#pragma unroll
        for (int jj = 0; jj < 8; ++jj) {
            a0[jj]     = s_probs[rowoff +  0 + kofs + jj];
            a0[8 + jj] = s_probs[rowoff +  8 + kofs + jj];
            a1[jj]     = s_probs[rowoff + 32 + kofs + jj];
            a1[8 + jj] = s_probs[rowoff + 40 + kofs + jj];
            a2[jj]     = s_probs[rowoff + 64 + kofs + jj];
            a2[8 + jj] = s_probs[rowoff + 72 + kofs + jj];
        }
    }

    // Each wave owns 96 output columns, processed as 3 blocks of 16x32.
    const size_t te_b = (size_t)b * D_ * L_;
    const size_t spanrow_base = (size_t)b * SPANS + (size_t)st * 16;

    for (int nt2 = 0; nt2 < 3; ++nt2) {
        const int ncb = wave * 96 + nt2 * 32;     // block's first column

#pragma unroll
        for (int h = 0; h < 2; ++h) {             // two 16-col subtiles
            const int col = ncb + h * 16 + l16;
            // E[t, col] = te[b, col, t] -> K-contiguous. B 32x16 layout:
            // lanes 0-15: K = kb+0..15, lanes 16-31: K = kb+16..31 (col = lane&15).
            const float* src = te + te_b + (size_t)col * L_ + hsel * 16;
            v16h b0 = load_b_frag(src);
            v16h b1 = load_b_frag(src + 32);
            v16h b2 = load_b_frag(src + 64);

            v8f c = {};
            c = __builtin_amdgcn_wmma_f32_16x16x32_f16(false, a0, false, b0, (short)0, c, false, false);
            c = __builtin_amdgcn_wmma_f32_16x16x32_f16(false, a1, false, b1, (short)0, c, false, false);
            c = __builtin_amdgcn_wmma_f32_16x16x32_f16(false, a2, false, b2, (short)0, c, false, false);

            // C/D layout: VGPR i -> row (i + 8*hsel), column = l16; stage to LDS.
#pragma unroll
            for (int i = 0; i < 8; ++i)
                s_stage[wave][i + 8 * hsel][h * 16 + l16] = c[i];
        }

        // Wave-local readback: each store instruction covers 4 rows x 128B contiguous.
#pragma unroll
        for (int q = 0; q < 4; ++q) {
            const int row  = q * 4 + (lane >> 3);
            const int colo = (lane & 7) * 4;
            float4 v = *(const float4*)&s_stage[wave][row][colo];
            *(float4*)&out[(spanrow_base + row) * (size_t)D_ + ncb + colo] = v;
        }
    }
}

extern "C" void kernel_launch(void* const* d_in, const int* in_sizes, int n_in,
                              void* d_out, int out_size, void* d_ws, size_t ws_size,
                              hipStream_t stream) {
    (void)in_sizes; (void)n_in; (void)out_size; (void)d_ws; (void)ws_size;
    const float* te  = (const float*)d_in[0];
    const int*   dep = (const int*)d_in[1];
    const int*   pos = (const int*)d_in[2];
    const int*   rel = (const int*)d_in[3];
    dim3 grid(SPANS / 16, B_, 1);   // 576 span tiles x 16 batches
    span_softhead_kernel<<<grid, 256, 0, stream>>>(te, dep, pos, rel, (float*)d_out);
}
